// CausalAttention_21045339750696
// MI455X (gfx1250) — matrix operand order repaired and checked by reading, not compiled, not run on hardware
//
#include <hip/hip_runtime.h>
#include <hip/hip_bf16.h>
#include <stdint.h>

typedef __bf16        v16bf __attribute__((ext_vector_type(16)));
typedef float         v8f   __attribute__((ext_vector_type(8)));
typedef unsigned int  v8u   __attribute__((ext_vector_type(8)));

#define BATCH   4
#define T_SEQ   2048
#define D_MODEL 2048
#define NH      16
#define HD      128

// ---------- small helpers ----------
static __device__ __forceinline__ unsigned short f2bf(float f) {
  unsigned int u = __builtin_bit_cast(unsigned int, f);
  u += 0x7FFFu + ((u >> 16) & 1u);          // round-to-nearest-even
  return (unsigned short)(u >> 16);
}
static __device__ __forceinline__ float bf2f(unsigned short h) {
  unsigned int u = ((unsigned int)h) << 16;
  return __builtin_bit_cast(float, u);
}
static __device__ __forceinline__ v8f zero8() {
  v8f z;
  #pragma unroll
  for (int i = 0; i < 8; ++i) z[i] = 0.0f;
  return z;
}

// A fragment (16x32 bf16, per ISA: lane half0 K{0..7,16..23}, half1 K{8..15,24..31}).
// Caller passes lane-resolved pointer (row*lda + hi*8); chunks at +0 and +16 elems.
static __device__ __forceinline__ v16bf load_frag_a(const unsigned short* p) {
  uint4 c0 = *(const uint4*)(p);
  uint4 c1 = *(const uint4*)(p + 16);
  v8u u;
  u[0] = c0.x; u[1] = c0.y; u[2] = c0.z; u[3] = c0.w;
  u[4] = c1.x; u[5] = c1.y; u[6] = c1.z; u[7] = c1.w;
  return __builtin_bit_cast(v16bf, u);
}
// B fragment (32x16 bf16): lane holds one column, half0 K{0..7,16..23}, half1 K{8..15,24..31}.
// Caller passes lane-resolved pointer (col_row*ldb + hi*8); chunks at +0 and +16 elems.
static __device__ __forceinline__ v16bf load_frag_b(const unsigned short* p) {
  uint4 c0 = *(const uint4*)(p);
  uint4 c1 = *(const uint4*)(p + 16);
  v8u u;
  u[0] = c0.x; u[1] = c0.y; u[2] = c0.z; u[3] = c0.w;
  u[4] = c1.x; u[5] = c1.y; u[6] = c1.z; u[7] = c1.w;
  return __builtin_bit_cast(v16bf, u);
}
static __device__ __forceinline__ v8f wmma_bf16(v16bf a, v16bf b, v8f c) {
  return __builtin_amdgcn_wmma_f32_16x16x32_bf16(false, a, false, b, (short)0, c,
                                                 false, false);
}

// ---------- f32 -> bf16 conversion (4 elems/thread, b128 loads) ----------
__global__ void cvt_bf16_kernel(const float* __restrict__ in,
                                unsigned short* __restrict__ out, int n4) {
  int i = blockIdx.x * blockDim.x + threadIdx.x;
  if (i >= n4) return;
  float4 f = ((const float4*)in)[i];
  uint2 p;
  p.x = (unsigned int)f2bf(f.x) | ((unsigned int)f2bf(f.y) << 16);
  p.y = (unsigned int)f2bf(f.z) | ((unsigned int)f2bf(f.w) << 16);
  ((uint2*)out)[i] = p;
}

// ---------- GEMM: C[M,N] = A[M,K] * W[N,K]^T, 64x64 tile per wave ----------
template <bool OUT_BF16>
__global__ void __launch_bounds__(32)
gemm_bf16_kernel(const unsigned short* __restrict__ A,
                 const unsigned short* __restrict__ Bw,
                 void* __restrict__ Cp, int M, int N, int K) {
  int lane = threadIdx.x;
  int hi = lane >> 4, lo = lane & 15;
  int n0 = blockIdx.x * 64;
  int m0 = blockIdx.y * 64;

  const unsigned short* ap[4];
  const unsigned short* bp[4];
  #pragma unroll
  for (int t = 0; t < 4; ++t) {
    ap[t] = A  + (size_t)(m0 + t * 16 + lo) * K + hi * 8;
    bp[t] = Bw + (size_t)(n0 + t * 16 + lo) * K + hi * 8;
  }

  v8f acc[4][4];
  #pragma unroll
  for (int ti = 0; ti < 4; ++ti)
    #pragma unroll
    for (int tj = 0; tj < 4; ++tj) acc[ti][tj] = zero8();

  for (int k0 = 0; k0 < K; k0 += 32) {
    v16bf af[4], bf_[4];
    #pragma unroll
    for (int t = 0; t < 4; ++t) af[t] = load_frag_a(ap[t] + k0);
    #pragma unroll
    for (int t = 0; t < 4; ++t) bf_[t] = load_frag_b(bp[t] + k0);
    #pragma unroll
    for (int ti = 0; ti < 4; ++ti)
      #pragma unroll
      for (int tj = 0; tj < 4; ++tj)
        acc[ti][tj] = wmma_bf16(af[ti], bf_[tj], acc[ti][tj]);
  }

  #pragma unroll
  for (int ti = 0; ti < 4; ++ti) {
    #pragma unroll
    for (int tj = 0; tj < 4; ++tj) {
      #pragma unroll
      for (int i = 0; i < 8; ++i) {
        int r = m0 + ti * 16 + hi * 8 + i;
        int c = n0 + tj * 16 + lo;
        if (OUT_BF16)
          ((unsigned short*)Cp)[(size_t)r * N + c] = f2bf(acc[ti][tj][i]);
        else
          ((float*)Cp)[(size_t)r * N + c] = acc[ti][tj][i];
      }
    }
  }
}

// ---------- RoPE + repack: qkv(bf16, [B*T,3D]) -> Q,K [B*H,T,HD], V^T [B*H,HD,T] ----------
__global__ void rope_pack_kernel(const unsigned short* __restrict__ qkv,
                                 const float* __restrict__ cosp,
                                 const float* __restrict__ sinp,
                                 unsigned short* __restrict__ Q,
                                 unsigned short* __restrict__ Ko,
                                 unsigned short* __restrict__ VT) {
  int idx = blockIdx.x * blockDim.x + threadIdx.x;   // B*T*H*64 threads
  int d2 = idx & 63;
  int h  = (idx >> 6) & (NH - 1);
  int t  = (idx >> 10) & (T_SEQ - 1);
  int b  = idx >> 21;
  if (b >= BATCH) return;

  size_t base = ((size_t)b * T_SEQ + t) * (3 * D_MODEL) + h * HD;
  float q1 = bf2f(qkv[base + d2]);
  float q2 = bf2f(qkv[base + d2 + 64]);
  float k1 = bf2f(qkv[base + D_MODEL + d2]);
  float k2 = bf2f(qkv[base + D_MODEL + d2 + 64]);
  unsigned short v1 = qkv[base + 2 * D_MODEL + d2];
  unsigned short v2 = qkv[base + 2 * D_MODEL + d2 + 64];

  float c = cosp[t * 64 + d2];
  float s = sinp[t * 64 + d2];
  const float scale = 0.08838834764831845f;  // 1/sqrt(128), folded into Q
  float qo1 = (q1 * c - q2 * s) * scale;
  float qo2 = (q2 * c + q1 * s) * scale;
  float ko1 = k1 * c - k2 * s;
  float ko2 = k2 * c + k1 * s;

  int bh = b * NH + h;
  size_t qb = ((size_t)bh * T_SEQ + t) * HD;
  Q[qb + d2]       = f2bf(qo1);
  Q[qb + d2 + 64]  = f2bf(qo2);
  Ko[qb + d2]      = f2bf(ko1);
  Ko[qb + d2 + 64] = f2bf(ko2);
  size_t vb = (size_t)bh * HD * T_SEQ;
  VT[vb + (size_t)d2 * T_SEQ + t]        = v1;
  VT[vb + (size_t)(d2 + 64) * T_SEQ + t] = v2;
}

// ---------- Flash attention: 1 wave per (b,h, 32-row Q tile); 32-key tiles ----------
// K/V fragments are loaded once and reused by both 16-row sub-tiles.
__global__ void __launch_bounds__(32)
flash_attn_kernel(const unsigned short* __restrict__ Q,
                  const unsigned short* __restrict__ Km,
                  const unsigned short* __restrict__ VT,
                  unsigned short* __restrict__ AO) {
  __shared__ __attribute__((aligned(16))) unsigned short pls[32 * 32];  // 2 KB

  int lane = threadIdx.x;
  int hi = lane >> 4, lo = lane & 15;
  int bid = blockIdx.x;
  int qt  = bid & (T_SEQ / 32 - 1);   // 64 tiles of 32 rows
  int bh  = bid >> 6;
  int b   = bh >> 4;
  int h   = bh & (NH - 1);
  int qt0 = qt * 32;

  const unsigned short* Qb = Q  + ((size_t)bh * T_SEQ + qt0) * HD;
  const unsigned short* Kb = Km + (size_t)bh * T_SEQ * HD;
  const unsigned short* Vb = VT + (size_t)bh * HD * T_SEQ;

  v16bf qf[2][4];
  #pragma unroll
  for (int r = 0; r < 2; ++r)
    #pragma unroll
    for (int c = 0; c < 4; ++c)
      qf[r][c] = load_frag_a(Qb + (r * 16 + lo) * HD + c * 32 + hi * 8);

  float mrow[2][8], lrow[2][8];
  #pragma unroll
  for (int r = 0; r < 2; ++r)
    #pragma unroll
    for (int i = 0; i < 8; ++i) { mrow[r][i] = -1e30f; lrow[r][i] = 0.0f; }
  v8f o[2][8];
  #pragma unroll
  for (int r = 0; r < 2; ++r)
    #pragma unroll
    for (int nt = 0; nt < 8; ++nt) o[r][nt] = zero8();

  for (int kt = 0; kt <= qt0 + 31; kt += 32) {
    // S[r] = Q[r] * K^T for keys [kt, kt+32); K frags shared across r
    v8f s[2][2];
    #pragma unroll
    for (int r = 0; r < 2; ++r) { s[r][0] = zero8(); s[r][1] = zero8(); }
    #pragma unroll
    for (int c = 0; c < 4; ++c) {
      v16bf kb0 = load_frag_b(Kb + (size_t)(kt + lo) * HD + c * 32 + hi * 8);
      v16bf kb1 = load_frag_b(Kb + (size_t)(kt + 16 + lo) * HD + c * 32 + hi * 8);
      #pragma unroll
      for (int r = 0; r < 2; ++r) {
        s[r][0] = wmma_bf16(qf[r][c], kb0, s[r][0]);
        s[r][1] = wmma_bf16(qf[r][c], kb1, s[r][1]);
      }
    }

    // online softmax per sub-tile (C-layout: row = qt0+r*16+hi*8+i, key = kt[+16]+lo)
    #pragma unroll
    for (int r = 0; r < 2; ++r) {
      float alpha[8];
      #pragma unroll
      for (int i = 0; i < 8; ++i) {
        int row = qt0 + r * 16 + hi * 8 + i;
        float a  = (kt + lo      <= row) ? s[r][0][i] : -1e30f;
        float bv = (kt + 16 + lo <= row) ? s[r][1][i] : -1e30f;
        float mx = fmaxf(a, bv);
        #pragma unroll
        for (int off = 1; off < 16; off <<= 1)
          mx = fmaxf(mx, __shfl_xor(mx, off, 32));
        float mn = fmaxf(mrow[r][i], mx);
        float al = __expf(mrow[r][i] - mn);
        float p0 = __expf(a - mn);
        float p1 = __expf(bv - mn);
        float rs = p0 + p1;
        #pragma unroll
        for (int off = 1; off < 16; off <<= 1)
          rs += __shfl_xor(rs, off, 32);
        lrow[r][i] = lrow[r][i] * al + rs;
        mrow[r][i] = mn;
        alpha[i] = al;
        pls[(r * 16 + hi * 8 + i) * 32 + lo]      = f2bf(p0);
        pls[(r * 16 + hi * 8 + i) * 32 + 16 + lo] = f2bf(p1);
      }
      #pragma unroll
      for (int nt = 0; nt < 8; ++nt)
        #pragma unroll
        for (int i = 0; i < 8; ++i) o[r][nt][i] *= alpha[i];
    }

    __syncthreads();                        // P visible in LDS
    v16bf pf0 = load_frag_a(&pls[lo * 32 + hi * 8]);
    v16bf pf1 = load_frag_a(&pls[(16 + lo) * 32 + hi * 8]);
    #pragma unroll
    for (int nt = 0; nt < 8; ++nt) {
      v16bf vbf = load_frag_b(Vb + (size_t)(nt * 16 + lo) * T_SEQ + kt + hi * 8);
      o[0][nt] = wmma_bf16(pf0, vbf, o[0][nt]);
      o[1][nt] = wmma_bf16(pf1, vbf, o[1][nt]);
    }
    __syncthreads();                        // LDS reads done before next overwrite
  }

  // normalize and scatter to attn_out (B,T,D) bf16
  #pragma unroll
  for (int r = 0; r < 2; ++r) {
    #pragma unroll
    for (int nt = 0; nt < 8; ++nt) {
      #pragma unroll
      for (int i = 0; i < 8; ++i) {
        int row = qt0 + r * 16 + hi * 8 + i;
        float v = o[r][nt][i] / lrow[r][i];
        AO[((size_t)b * T_SEQ + row) * D_MODEL + h * HD + nt * 16 + lo] = f2bf(v);
      }
    }
  }
}

// ---------- host orchestration ----------
extern "C" void kernel_launch(void* const* d_in, const int* in_sizes, int n_in,
                              void* d_out, int out_size, void* d_ws, size_t ws_size,
                              hipStream_t stream) {
  (void)in_sizes; (void)n_in; (void)out_size; (void)ws_size;
  const float* x     = (const float*)d_in[0];
  const float* cosp  = (const float*)d_in[1];
  const float* sinp  = (const float*)d_in[2];
  const float* W_qkv = (const float*)d_in[3];
  const float* W_out = (const float*)d_in[4];
  float* out = (float*)d_out;

  char* ws = (char*)d_ws;
  size_t off = 0;
  auto carve = [&](size_t bytes) -> void* {
    void* p = ws + off;
    off += (bytes + 255) & ~(size_t)255;
    return p;
  };
  const size_t BT = (size_t)BATCH * T_SEQ;
  unsigned short* xb   = (unsigned short*)carve(BT * D_MODEL * 2);
  unsigned short* wqb  = (unsigned short*)carve((size_t)3 * D_MODEL * D_MODEL * 2);
  unsigned short* wob  = (unsigned short*)carve((size_t)D_MODEL * D_MODEL * 2);
  unsigned short* qkvb = (unsigned short*)carve(BT * 3 * D_MODEL * 2);
  unsigned short* Qb   = (unsigned short*)carve(BT * D_MODEL * 2);
  unsigned short* Kb   = (unsigned short*)carve(BT * D_MODEL * 2);
  unsigned short* VTb  = (unsigned short*)carve(BT * D_MODEL * 2);
  unsigned short* AOb  = (unsigned short*)carve(BT * D_MODEL * 2);

  int nx4 = (int)(BT * D_MODEL / 4);
  cvt_bf16_kernel<<<(nx4 + 255) / 256, 256, 0, stream>>>(x, xb, nx4);
  int nwq4 = 3 * D_MODEL * D_MODEL / 4;
  cvt_bf16_kernel<<<(nwq4 + 255) / 256, 256, 0, stream>>>(W_qkv, wqb, nwq4);
  int nwo4 = D_MODEL * D_MODEL / 4;
  cvt_bf16_kernel<<<(nwo4 + 255) / 256, 256, 0, stream>>>(W_out, wob, nwo4);

  // qkv = x @ W_qkv^T   (M=B*T, N=3D, K=D), bf16 out
  dim3 g1(3 * D_MODEL / 64, (unsigned)(BT / 64));
  gemm_bf16_kernel<true><<<g1, 32, 0, stream>>>(xb, wqb, (void*)qkvb,
                                                (int)BT, 3 * D_MODEL, D_MODEL);

  int nr = (int)(BT * NH * 64);
  rope_pack_kernel<<<(nr + 255) / 256, 256, 0, stream>>>(qkvb, cosp, sinp,
                                                         Qb, Kb, VTb);

  flash_attn_kernel<<<BATCH * NH * (T_SEQ / 32), 32, 0, stream>>>(Qb, Kb, VTb, AOb);

  // out = attn_out @ W_out^T  (f32 out)
  dim3 g2(D_MODEL / 64, (unsigned)(BT / 64));
  gemm_bf16_kernel<false><<<g2, 32, 0, stream>>>(AOb, wob, (void*)out,
                                                 (int)BT, D_MODEL, D_MODEL);
}
